// PairformerLayer_515396076361
// MI455X (gfx1250) — compile-verified, matrix operand order checked by reading, not run on hardware
//
#include <hip/hip_runtime.h>
#include <hip/hip_bf16.h>
#include <stdint.h>

typedef __bf16 bf16;
typedef __attribute__((ext_vector_type(16))) __bf16 v16bf;
typedef __attribute__((ext_vector_type(8)))  float  v8f;

#define NSEQ 256
static const long NNL = 65536L;

__device__ __forceinline__ float sigf(float x) { return 1.f / (1.f + __expf(-x)); }

union BV { v16bf v; uint4 q[2]; };

// ---------------------------------------------------------------------------
// Generic bf16 WMMA GEMM:  C[M,Ncols] = act(A[M,K] * W[K,Ncols] + bias)
// W is pre-transposed (and zero padded) into Wt[Npad][K] row-major bf16.
// Register-blocked 2x2: each wave computes a 32x32 output block (4 WMMAs per
// 32-wide K step, 8x 16-byte loads), 8 waves/block -> 64x128 block tile.
// Optional batching (strideA/B/C in elements) and generalized C strides
// (ldcm per row, ldcn per col) so the triangle einsum can write channel-last.
// ACT: 0 none, 1 sigmoid.  OUT: 0 f32, 1 bf16.
// ---------------------------------------------------------------------------
template <int ACT, int OUT>
__global__ __launch_bounds__(256) void gemm_bf16_wmma(
    const bf16* __restrict__ A, const bf16* __restrict__ Wt,
    float* __restrict__ Cf, bf16* __restrict__ Cb, const float* __restrict__ bias,
    int M, int Ncols, int Npad, int K, int lda, long ldcm, int ldcn,
    long strideA, long strideB, long strideC)
{
  int lane = threadIdx.x & 31, wv = threadIdx.x >> 5;
  int wm = wv & 1, wn = wv >> 1;                 // 2 x 4 wave grid
  int mrow = blockIdx.y * 64 + wm * 32;
  int nb   = blockIdx.x * 128 + wn * 32;
  if (mrow >= M || nb >= Npad) return;
  bool hasN1 = (nb + 16) < Npad;
  long zb = blockIdx.z;
  A  += zb * strideA;
  Wt += zb * strideB;
  int nl = lane & 15, half = lane >> 4;

  const bf16* pa0 = A + (long)(mrow + nl) * lda + half * 8;
  const bf16* pa1 = pa0 + 16L * lda;
  const bf16* pb0 = Wt + (long)(nb + nl) * K + half * 8;
  const bf16* pb1 = pb0 + 16L * K;

  v8f a00 = {}, a01 = {}, a10 = {}, a11 = {};
  for (int kb = 0; kb < K; kb += 32) {
    // prefetch next K step into the cache hierarchy (global_prefetch_b8)
    if (kb + 32 < K) {
      __builtin_prefetch(pa0 + kb + 32, 0, 1);
      __builtin_prefetch(pb0 + kb + 32, 0, 1);
    }
    BV fa0, fa1, fb0, fb1;
    fa0.q[0] = *(const uint4*)(pa0 + kb);
    fa0.q[1] = *(const uint4*)(pa0 + kb + 16);
    fa1.q[0] = *(const uint4*)(pa1 + kb);
    fa1.q[1] = *(const uint4*)(pa1 + kb + 16);
    fb0.q[0] = *(const uint4*)(pb0 + kb);
    fb0.q[1] = *(const uint4*)(pb0 + kb + 16);
    a00 = __builtin_amdgcn_wmma_f32_16x16x32_bf16(false, fa0.v, false, fb0.v, (short)0, a00, false, false);
    a10 = __builtin_amdgcn_wmma_f32_16x16x32_bf16(false, fa1.v, false, fb0.v, (short)0, a10, false, false);
    if (hasN1) {
      fb1.q[0] = *(const uint4*)(pb1 + kb);
      fb1.q[1] = *(const uint4*)(pb1 + kb + 16);
      a01 = __builtin_amdgcn_wmma_f32_16x16x32_bf16(false, fa0.v, false, fb1.v, (short)0, a01, false, false);
      a11 = __builtin_amdgcn_wmma_f32_16x16x32_bf16(false, fa1.v, false, fb1.v, (short)0, a11, false, false);
    }
  }

  long cbase = zb * strideC;
  auto store_tile = [&](v8f& acc, int mt, int nt) {
    int col = nt + nl;
    if (col >= Ncols) return;
    float bval = bias ? bias[col] : 0.f;
#pragma unroll
    for (int r = 0; r < 8; ++r) {
      int row = mt + r + half * 8;
      float v = acc[r] + bval;
      if (ACT == 1) v = sigf(v);
      long addr = cbase + (long)row * ldcm + (long)col * ldcn;
      if (OUT == 0) Cf[addr] = v;
      else          Cb[addr] = (bf16)v;
    }
  };
  store_tile(a00, mrow, nb);
  store_tile(a10, mrow + 16, nb);
  if (hasN1) {
    store_tile(a01, mrow, nb + 16);
    store_tile(a11, mrow + 16, nb + 16);
  }
}

// ---------------------------------------------------------------------------
// LayerNorm over last dim C (128 or 384). One block (128 thr) per row.
// Optional transposed read over an Ndim x Ndim grid (for tri-attn "ending").
// ---------------------------------------------------------------------------
__global__ __launch_bounds__(128) void ln_kernel(
    const float* __restrict__ x, const float* __restrict__ w, const float* __restrict__ b,
    float* __restrict__ outf, bf16* __restrict__ outbf, int C, int transposed, int Ndim)
{
  long r = blockIdx.x;
  long src = r;
  if (transposed) { long i = r / Ndim, j = r - i * Ndim; src = j * (long)Ndim + i; }
  const float* xr = x + src * C;
  int t = threadIdx.x, lane = t & 31, wv = t >> 5;
  int per = C >> 7;
  float vals[3];
  float s = 0.f;
  for (int e = 0; e < per; ++e) { vals[e] = xr[t + 128 * e]; s += vals[e]; }
  __shared__ float sm[8];
  for (int off = 16; off; off >>= 1) s += __shfl_xor(s, off, 32);
  if (!lane) sm[wv] = s;
  __syncthreads();
  s = sm[0] + sm[1] + sm[2] + sm[3];
  float mean = s / C;
  float v2 = 0.f;
  for (int e = 0; e < per; ++e) { float d = vals[e] - mean; v2 += d * d; }
  for (int off = 16; off; off >>= 1) v2 += __shfl_xor(v2, off, 32);
  if (!lane) sm[4 + wv] = v2;
  __syncthreads();
  v2 = sm[4] + sm[5] + sm[6] + sm[7];
  float rstd = rsqrtf(v2 / C + 1e-5f);
  for (int e = 0; e < per; ++e) {
    int c = t + 128 * e;
    float o = (vals[e] - mean) * rstd * w[c] + b[c];
    if (outf)  outf[r * C + c] = o;
    if (outbf) outbf[r * C + c] = (bf16)o;
  }
}

// ---------------------------------------------------------------------------
// Flash attention (rows x 256 keys), head dim padded to 32, WMMA bf16.
// One wave = 16 query rows; 8 waves/block = 128 rows; grid(2, H, batch).
// q/k/v/o layout: [batch][row][h*32+d]; bias layout [row][col][H] f32.
// ---------------------------------------------------------------------------
__global__ __launch_bounds__(256) void attn_kernel(
    const bf16* __restrict__ q, const bf16* __restrict__ k, const bf16* __restrict__ v,
    const float* __restrict__ bias, float* __restrict__ o,
    int HD, int H, int validD, float scale, long batchStride)
{
  __shared__ __align__(16) bf16 VT[32][264];        // V transposed: [d][key]
  __shared__ __align__(16) bf16 Pst[8][16][32];     // per-wave P staging
  const int N = 256;
  int h = blockIdx.y;
  long zb = (long)blockIdx.z * batchStride;
  const bf16* qb = q + zb + h * 32;
  const bf16* kb = k + zb + h * 32;
  const bf16* vb = v + zb + h * 32;

  for (int idx = threadIdx.x; idx < 32 * N; idx += 256) {
    int d = idx >> 8, kk = idx & 255;
    VT[d][kk] = vb[(long)kk * HD + d];
  }
  __syncthreads();

  int lane = threadIdx.x & 31, wv = threadIdx.x >> 5;
  int row0 = blockIdx.x * 128 + wv * 16;
  int cn = lane & 15, half = lane >> 4;

  BV Qa;
  const bf16* qp = qb + (long)(row0 + cn) * HD + half * 8;
  Qa.q[0] = *(const uint4*)qp;
  Qa.q[1] = *(const uint4*)(qp + 16);

  float mi[8], li[8];
#pragma unroll
  for (int r = 0; r < 8; ++r) { mi[r] = -1e30f; li[r] = 0.f; }
  v8f o1 = {}, o2 = {};
  v8f zc = {};

  for (int k0 = 0; k0 < N; k0 += 32) {
    BV K1, K2;
    const bf16* kp1 = kb + (long)(k0 + cn) * HD + half * 8;
    const bf16* kp2 = kp1 + 16 * HD;
    K1.q[0] = *(const uint4*)kp1; K1.q[1] = *(const uint4*)(kp1 + 16);
    K2.q[0] = *(const uint4*)kp2; K2.q[1] = *(const uint4*)(kp2 + 16);

    v8f s1 = __builtin_amdgcn_wmma_f32_16x16x32_bf16(false, Qa.v, false, K1.v, (short)0, zc, false, false);
    v8f s2 = __builtin_amdgcn_wmma_f32_16x16x32_bf16(false, Qa.v, false, K2.v, (short)0, zc, false, false);

#pragma unroll
    for (int r = 0; r < 8; ++r) {
      long row = row0 + r + half * 8;
      float b1 = bias[row * (long)(N * H) + (long)(k0 + cn) * H + h];
      float b2 = bias[row * (long)(N * H) + (long)(k0 + 16 + cn) * H + h];
      s1[r] = s1[r] * scale + b1;
      s2[r] = s2[r] * scale + b2;
      float mx = fmaxf(s1[r], s2[r]);
      for (int off = 1; off < 16; off <<= 1) mx = fmaxf(mx, __shfl_xor(mx, off, 32));
      float mn = fmaxf(mi[r], mx);
      float corr = __expf(mi[r] - mn);
      float p1 = __expf(s1[r] - mn);
      float p2 = __expf(s2[r] - mn);
      float rs = p1 + p2;
      for (int off = 1; off < 16; off <<= 1) rs += __shfl_xor(rs, off, 32);
      li[r] = li[r] * corr + rs;
      mi[r] = mn;
      o1[r] *= corr;
      o2[r] *= corr;
      Pst[wv][r + half * 8][cn] = (bf16)p1;
      Pst[wv][r + half * 8][16 + cn] = (bf16)p2;
    }

    BV Pa, V1, V2;
    Pa.q[0] = *(const uint4*)&Pst[wv][cn][half * 8];
    Pa.q[1] = *(const uint4*)&Pst[wv][cn][half * 8 + 16];
    V1.q[0] = *(const uint4*)&VT[cn][k0 + half * 8];
    V1.q[1] = *(const uint4*)&VT[cn][k0 + half * 8 + 16];
    V2.q[0] = *(const uint4*)&VT[16 + cn][k0 + half * 8];
    V2.q[1] = *(const uint4*)&VT[16 + cn][k0 + half * 8 + 16];

    o1 = __builtin_amdgcn_wmma_f32_16x16x32_bf16(false, Pa.v, false, V1.v, (short)0, o1, false, false);
    o2 = __builtin_amdgcn_wmma_f32_16x16x32_bf16(false, Pa.v, false, V2.v, (short)0, o2, false, false);
  }

#pragma unroll
  for (int r = 0; r < 8; ++r) {
    int row = row0 + r + half * 8;
    float inv = 1.f / li[r];
    long ob = zb + (long)row * HD + h * 32;
    o[ob + cn] = o1[r] * inv;
    int c2 = 16 + cn;
    if (c2 < validD) o[ob + c2] = o2[r] * inv;
  }
}

// --------------------------- elementwise kernels ---------------------------
__global__ void k_copy(float* d, const float* s, long n) {
  long i = (long)blockIdx.x * 256 + threadIdx.x; if (i < n) d[i] = s[i];
}
__global__ void k_zero(float* d, long n) {
  long i = (long)blockIdx.x * 256 + threadIdx.x; if (i < n) d[i] = 0.f;
}
__global__ void k_to_bf(bf16* d, const float* s, long n) {
  long i = (long)blockIdx.x * 256 + threadIdx.x; if (i < n) d[i] = (bf16)s[i];
}
__global__ void k_add(float* d, const float* s, long n) {
  long i = (long)blockIdx.x * 256 + threadIdx.x; if (i < n) d[i] += s[i];
}
__global__ void k_add_tr(float* z, const float* x, int Nd, int C) {
  long i = (long)blockIdx.x * 256 + threadIdx.x;
  long n = (long)Nd * Nd * C;
  if (i < n) {
    int c = i % C;
    long t = i / C;
    int jj = t % Nd, ii = t / Nd;
    z[i] += x[((long)jj * Nd + ii) * C + c];
  }
}
__global__ void k_fma(float* d, const float* a, const float* b, long n) {
  long i = (long)blockIdx.x * 256 + threadIdx.x; if (i < n) d[i] += a[i] * b[i];
}
__global__ void k_mul_bf(bf16* d, const float* a, const float* b, long n) {
  long i = (long)blockIdx.x * 256 + threadIdx.x; if (i < n) d[i] = (bf16)(a[i] * b[i]);
}
__global__ void k_silu_mul_bf(bf16* d, const float* a, const float* b, long n) {
  long i = (long)blockIdx.x * 256 + threadIdx.x;
  if (i < n) { float x = a[i]; d[i] = (bf16)(x * sigf(x) * b[i]); }
}
// ab = P * sigmoid(G); split into a/b halves, write channel-major (c,i,k) bf16,
// optionally swapping the (i0,i1) grid axes (incoming triangle mult).
__global__ void k_make_abT(const float* P, const float* G, bf16* aT, bf16* bT,
                           int Nd, int swap) {
  long i = (long)blockIdx.x * 256 + threadIdx.x;
  long n = (long)Nd * Nd * 128;
  if (i >= n) return;
  int c = i & 127;
  long t = i >> 7;
  int i1 = t % Nd, i0 = t / Nd;
  long rb = t * 256;
  float va = P[rb + c] * sigf(G[rb + c]);
  float vb = P[rb + 128 + c] * sigf(G[rb + 128 + c]);
  long dst = (long)c * Nd * Nd + (swap ? ((long)i1 * Nd + i0) : t);
  aT[dst] = (bf16)va;
  bT[dst] = (bf16)vb;
}
// weight prep: f32 [K][Ncols] -> bf16 [Npad][K] (transposed, zero-padded cols)
__global__ void k_wprep(const float* w, bf16* wt, int K, int Ncols, int Npad) {
  long i = (long)blockIdx.x * 256 + threadIdx.x;
  long n = (long)Npad * K;
  if (i >= n) return;
  int kk = i % K, np = i / K;
  wt[i] = (np < Ncols) ? (bf16)w[(long)kk * Ncols + np] : (bf16)0.f;
}
// apb q/k/v/g: [384][h*24+d] -> [512][384] with d padded 24->32
__global__ void k_wprep_colpad(const float* w, bf16* wt, int K) {
  long i = (long)blockIdx.x * 256 + threadIdx.x;
  long n = 512L * K;
  if (i >= n) return;
  int kk = i % K, np = i / K, h = np >> 5, d = np & 31;
  wt[i] = (d < 24) ? (bf16)w[(long)kk * 384 + h * 24 + d] : (bf16)0.f;
}
// apb_o: [h*24+d][384] -> [384][512] with padded K rows zeroed
__global__ void k_wprep_rowpad(const float* w, bf16* wt) {
  long i = (long)blockIdx.x * 256 + threadIdx.x;
  long n = 384L * 512;
  if (i >= n) return;
  int kp = i % 512, np = i / 512, h = kp >> 5, d = kp & 31;
  wt[i] = (d < 24) ? (bf16)w[(long)(h * 24 + d) * 384 + np] : (bf16)0.f;
}
__global__ void k_qbprep(const float* qb, float* out) {
  int i = blockIdx.x * 256 + threadIdx.x;
  if (i < 512) { int h = i >> 5, d = i & 31; out[i] = (d < 24) ? qb[h * 24 + d] : 0.f; }
}

// ---------------------------------------------------------------------------
extern "C" void kernel_launch(void* const* d_in, const int* in_sizes, int n_in,
                              void* d_out, int out_size, void* d_ws, size_t ws_size,
                              hipStream_t stream) {
  (void)in_sizes; (void)n_in; (void)out_size; (void)ws_size;
  const int N = NSEQ;
  const long NNl = NNL;

  const float* s_in = (const float*)d_in[0];
  const float* z_in = (const float*)d_in[1];
  auto in = [&](int i) { return (const float*)d_in[i]; };

  // workspace layout
  char* ws = (char*)d_ws;
  const size_t O_WS0 = 0;                       // 32MB  f32 scratch / bf16 s
  const size_t O_WS1 = O_WS0 + 33554432;        // 16MB  bf16 (zn / gated)
  const size_t O_WSA = O_WS1 + 16777216;        // 128MB f32 bigA
  const size_t O_WSB = O_WSA + 134217728;       // 128MB f32 bigB
  const size_t O_WS4 = O_WSB + 134217728;       // 72MB  bf16/f32 region
  const size_t O_WS5 = O_WS4 + 75497472;        // 32MB  f32
  const size_t O_WSW = O_WS5 + 33554432;        // 16MB  prepped weights

  float* WS0f = (float*)(ws + O_WS0);
  bf16*  WS1b = (bf16*)(ws + O_WS1);
  float* bigA = (float*)(ws + O_WSA);
  float* bigB = (float*)(ws + O_WSB);
  char*  WS4  = ws + O_WS4;
  float* WS5f = (float*)(ws + O_WS5);
  bf16*  Wb   = (bf16*)(ws + O_WSW);
  float* qbpad = (float*)(ws + O_WSW + 15728640);

  float* sOut = (float*)d_out;                  // 256*384
  float* zOut = (float*)d_out + 98304;          // 256*256*128

  auto ew = [&](long n) { return dim3((unsigned)((n + 255) / 256)); };

  // ---- prep weights ----
  long wofs = 0;
  auto WAL = [&](long e) { long o = wofs; wofs += e; return Wb + o; };
  auto prep = [&](const float* w, bf16* dst, int K, int Nc, int Np) {
    long n = (long)Np * K;
    k_wprep<<<ew(n), 256, 0, stream>>>(w, dst, K, Nc, Np);
  };

  struct TM { bf16 *pin, *gin, *pout, *gout; };
  struct TA { bf16 *q, *k, *v, *b, *g, *o; };
  TM tm[2]; TA ta[2];
  for (int p = 0; p < 2; ++p) {
    int base = 2 + p * 8;
    tm[p].pin  = WAL(256L * 128); prep(in(base + 2), tm[p].pin, 128, 256, 256);
    tm[p].gin  = WAL(256L * 128); prep(in(base + 3), tm[p].gin, 128, 256, 256);
    tm[p].pout = WAL(128L * 128); prep(in(base + 6), tm[p].pout, 128, 128, 128);
    tm[p].gout = WAL(128L * 128); prep(in(base + 7), tm[p].gout, 128, 128, 128);
  }
  for (int p = 0; p < 2; ++p) {
    int base = 18 + p * 8;
    ta[p].q = WAL(128L * 128); prep(in(base + 2), ta[p].q, 128, 128, 128);
    ta[p].k = WAL(128L * 128); prep(in(base + 3), ta[p].k, 128, 128, 128);
    ta[p].v = WAL(128L * 128); prep(in(base + 4), ta[p].v, 128, 128, 128);
    ta[p].b = WAL(16L * 128);  prep(in(base + 5), ta[p].b, 128, 4, 16);
    ta[p].g = WAL(128L * 128); prep(in(base + 6), ta[p].g, 128, 128, 128);
    ta[p].o = WAL(128L * 128); prep(in(base + 7), ta[p].o, 128, 128, 128);
  }
  bf16* apbq = WAL(512L * 384); k_wprep_colpad<<<ew(512L * 384), 256, 0, stream>>>(in(34), apbq, 384);
  bf16* apbk = WAL(512L * 384); k_wprep_colpad<<<ew(512L * 384), 256, 0, stream>>>(in(36), apbk, 384);
  bf16* apbv = WAL(512L * 384); k_wprep_colpad<<<ew(512L * 384), 256, 0, stream>>>(in(37), apbv, 384);
  bf16* apbg = WAL(512L * 384); k_wprep_colpad<<<ew(512L * 384), 256, 0, stream>>>(in(38), apbg, 384);
  bf16* apbz = WAL(16L * 128);  prep(in(41), apbz, 128, 16, 16);
  bf16* apbo = WAL(384L * 512); k_wprep_rowpad<<<ew(384L * 512), 256, 0, stream>>>(in(42), apbo);
  k_qbprep<<<2, 256, 0, stream>>>(in(35), qbpad);
  bf16* tzf1 = WAL(512L * 128); prep(in(45), tzf1, 128, 512, 512);
  bf16* tzf2 = WAL(512L * 128); prep(in(46), tzf2, 128, 512, 512);
  bf16* tzf3 = WAL(128L * 512); prep(in(47), tzf3, 512, 128, 128);
  bf16* tsf1 = WAL(1536L * 384); prep(in(50), tsf1, 384, 1536, 1536);
  bf16* tsf2 = WAL(1536L * 384); prep(in(51), tsf2, 384, 1536, 1536);
  bf16* tsf3 = WAL(384L * 1536); prep(in(52), tsf3, 1536, 384, 384);

  auto gemm = [&](int act, const bf16* A, const bf16* Wt, void* Cout, int outbf,
                  const float* bias, long M, int Ncols, int Npad, int K, int lda,
                  long ldcm, int ldcn, int batch, long sA, long sB, long sC) {
    dim3 g((unsigned)((Npad + 127) / 128), (unsigned)((M + 63) / 64), (unsigned)batch);
    if (act == 1)
      gemm_bf16_wmma<1, 0><<<g, 256, 0, stream>>>(A, Wt, (float*)Cout, nullptr, bias,
          (int)M, Ncols, Npad, K, lda, ldcm, ldcn, sA, sB, sC);
    else if (outbf)
      gemm_bf16_wmma<0, 1><<<g, 256, 0, stream>>>(A, Wt, nullptr, (bf16*)Cout, bias,
          (int)M, Ncols, Npad, K, lda, ldcm, ldcn, sA, sB, sC);
    else
      gemm_bf16_wmma<0, 0><<<g, 256, 0, stream>>>(A, Wt, (float*)Cout, nullptr, bias,
          (int)M, Ncols, Npad, K, lda, ldcm, ldcn, sA, sB, sC);
  };

  // running copies of s and z
  k_copy<<<ew(98304), 256, 0, stream>>>(sOut, s_in, 98304);
  k_copy<<<ew(NNl * 128), 256, 0, stream>>>(zOut, z_in, NNl * 128);

  bf16* A4_0 = (bf16*)(WS4);
  bf16* A4_1 = (bf16*)(WS4 + 16777216);
  bf16* A4_2 = (bf16*)(WS4 + 33554432);
  float* biasBuf = (float*)(WS4 + 50331648);

  // -------------------- triangle multiplication ----------------------------
  auto tri_mul = [&](const TM& W, const float* nw, const float* nb,
                     const float* now, const float* nob, int outgoing) {
    ln_kernel<<<(unsigned)NNl, 128, 0, stream>>>(zOut, nw, nb, nullptr, WS1b, 128, 0, N);
    gemm(0, WS1b, W.pin, bigA, 0, nullptr, NNl, 256, 256, 128, 128, 256, 1, 1, 0, 0, 0);
    gemm(0, WS1b, W.gin, bigB, 0, nullptr, NNl, 256, 256, 128, 128, 256, 1, 1, 0, 0, 0);
    k_make_abT<<<ew(NNl * 128), 256, 0, stream>>>(bigA, bigB, A4_0, A4_1, N, outgoing ? 0 : 1);
    gemm(1, WS1b, W.gout, bigA, 0, nullptr, NNl, 128, 128, 128, 128, 128, 1, 1, 0, 0, 0);
    // batched per-channel GEMM: x[i,j,c] = sum_k a[c][i][k]*b[c][j][k]
    gemm(0, A4_0, A4_1, WS5f, 0, nullptr, N, 256, 256, 256, 256,
         (long)N * 128, 128, 128, NNl, NNl, 1);
    ln_kernel<<<(unsigned)NNl, 128, 0, stream>>>(WS5f, now, nob, nullptr, WS1b, 128, 0, N);
    gemm(0, WS1b, W.pout, bigB, 0, nullptr, NNl, 128, 128, 128, 128, 128, 1, 1, 0, 0, 0);
    k_fma<<<ew(NNl * 128), 256, 0, stream>>>(zOut, bigA, bigB, NNl * 128);
  };
  tri_mul(tm[0], in(2), in(3), in(6), in(7), 1);
  tri_mul(tm[1], in(10), in(11), in(14), in(15), 0);

  // -------------------- triangle attention ---------------------------------
  auto tri_attn = [&](const TA& W, const float* nw, const float* nb, int ending) {
    ln_kernel<<<(unsigned)NNl, 128, 0, stream>>>(zOut, nw, nb, nullptr, WS1b, 128, ending, N);
    gemm(0, WS1b, W.q, A4_0, 1, nullptr, NNl, 128, 128, 128, 128, 128, 1, 1, 0, 0, 0);
    gemm(0, WS1b, W.k, A4_1, 1, nullptr, NNl, 128, 128, 128, 128, 128, 1, 1, 0, 0, 0);
    gemm(0, WS1b, W.v, A4_2, 1, nullptr, NNl, 128, 128, 128, 128, 128, 1, 1, 0, 0, 0);
    gemm(0, WS1b, W.b, biasBuf, 0, nullptr, NNl, 4, 16, 128, 128, 4, 1, 1, 0, 0, 0);
    gemm(1, WS1b, W.g, bigA, 0, nullptr, NNl, 128, 128, 128, 128, 128, 1, 1, 0, 0, 0);
    attn_kernel<<<dim3(2, 4, 256), 256, 0, stream>>>(A4_0, A4_1, A4_2, biasBuf, WS5f,
        128, 4, 32, 0.1767766953f, (long)N * 128);
    k_mul_bf<<<ew(NNl * 128), 256, 0, stream>>>(WS1b, WS5f, bigA, NNl * 128);
    gemm(0, WS1b, W.o, bigB, 0, nullptr, NNl, 128, 128, 128, 128, 128, 1, 1, 0, 0, 0);
    if (ending) k_add_tr<<<ew(NNl * 128), 256, 0, stream>>>(zOut, bigB, N, 128);
    else        k_add<<<ew(NNl * 128), 256, 0, stream>>>(zOut, bigB, NNl * 128);
  };
  tri_attn(ta[0], in(18), in(19), 0);
  tri_attn(ta[1], in(26), in(27), 1);

  // -------------------- z transition (SwiGLU) ------------------------------
  ln_kernel<<<(unsigned)NNl, 128, 0, stream>>>(zOut, in(43), in(44), nullptr, WS1b, 128, 0, N);
  gemm(0, WS1b, tzf1, bigA, 0, nullptr, NNl, 512, 512, 128, 128, 512, 1, 1, 0, 0, 0);
  gemm(0, WS1b, tzf2, bigB, 0, nullptr, NNl, 512, 512, 128, 128, 512, 1, 1, 0, 0, 0);
  k_silu_mul_bf<<<ew(NNl * 512), 256, 0, stream>>>((bf16*)WS4, bigA, bigB, NNl * 512);
  gemm(0, (bf16*)WS4, tzf3, WS5f, 0, nullptr, NNl, 128, 128, 512, 512, 128, 1, 1, 0, 0, 0);
  k_add<<<ew(NNl * 128), 256, 0, stream>>>(zOut, WS5f, NNl * 128);

  // -------------------- attention with pair bias on s ----------------------
  ln_kernel<<<(unsigned)NNl, 128, 0, stream>>>(zOut, in(39), in(40), nullptr, WS1b, 128, 0, N);
  gemm(0, WS1b, apbz, biasBuf, 0, nullptr, NNl, 16, 16, 128, 128, 16, 1, 1, 0, 0, 0);
  bf16* sbf = (bf16*)WS0f;
  k_to_bf<<<ew(98304), 256, 0, stream>>>(sbf, sOut, 98304);
  gemm(0, sbf, apbq, A4_0, 1, qbpad, 256, 512, 512, 384, 384, 512, 1, 1, 0, 0, 0);
  gemm(0, sbf, apbk, A4_1, 1, nullptr, 256, 512, 512, 384, 384, 512, 1, 1, 0, 0, 0);
  gemm(0, sbf, apbv, A4_2, 1, nullptr, 256, 512, 512, 384, 384, 512, 1, 1, 0, 0, 0);
  gemm(1, sbf, apbg, bigA, 0, nullptr, 256, 512, 512, 384, 384, 512, 1, 1, 0, 0, 0);
  k_zero<<<ew(256L * 512), 256, 0, stream>>>(bigB, 256L * 512);
  attn_kernel<<<dim3(2, 16, 1), 256, 0, stream>>>(A4_0, A4_1, A4_2, biasBuf, bigB,
      512, 16, 24, 0.2041241452f, 0);
  k_mul_bf<<<ew(256L * 512), 256, 0, stream>>>(WS1b, bigB, bigA, 256L * 512);
  gemm(0, WS1b, apbo, WS5f, 0, nullptr, 256, 384, 384, 512, 512, 384, 1, 1, 0, 0, 0);
  k_add<<<ew(98304), 256, 0, stream>>>(sOut, WS5f, 98304);

  // -------------------- s transition (SwiGLU) ------------------------------
  ln_kernel<<<256, 128, 0, stream>>>(sOut, in(48), in(49), nullptr, WS1b, 384, 0, N);
  gemm(0, WS1b, tsf1, bigA, 0, nullptr, 256, 1536, 1536, 384, 384, 1536, 1, 1, 0, 0, 0);
  gemm(0, WS1b, tsf2, bigB, 0, nullptr, 256, 1536, 1536, 384, 384, 1536, 1, 1, 0, 0, 0);
  k_silu_mul_bf<<<ew(256L * 1536), 256, 0, stream>>>((bf16*)WS4, bigA, bigB, 256L * 1536);
  gemm(0, (bf16*)WS4, tsf3, WS5f, 0, nullptr, 256, 384, 384, 1536, 1536, 384, 1, 1, 0, 0, 0);
  k_add<<<ew(98304), 256, 0, stream>>>(sOut, WS5f, 98304);
}
